// DGCNNCls_55722905698547
// MI455X (gfx1250) — matrix operand order repaired and checked by reading, not compiled, not run on hardware
//
#include <hip/hip_runtime.h>
#include <hip/hip_bf16.h>

typedef __attribute__((ext_vector_type(16))) _Float16 v16h;
typedef __attribute__((ext_vector_type(8)))  float    v8f;
typedef int gv4i __attribute__((__vector_size__(4 * sizeof(int))));

#define B_   32
#define N_   1024
#define K_   20
#define P_   (B_ * N_)      // 32768 points
#define E_   (P_ * K_)      // 655360 edge rows
#define CATD 320

#ifndef __has_builtin
#define __has_builtin(x) 0
#endif

#if __has_builtin(__builtin_amdgcn_global_load_async_to_lds_b128)
#define HAVE_ASYNC_B128 1
#else
#define HAVE_ASYNC_B128 0
#endif

#define AS3 __attribute__((address_space(3)))

__device__ __forceinline__ AS3 gv4i* lds_gv4i_ptr(void* p) {
  // generic LDS addresses keep the LDS byte offset in the low 32 bits
  return (AS3 gv4i*)(unsigned)(uintptr_t)p;
}

__device__ __forceinline__ void wait_async0() {
#if __has_builtin(__builtin_amdgcn_s_wait_asynccnt)
  __builtin_amdgcn_s_wait_asynccnt(0);
#else
  asm volatile("s_wait_asynccnt 0" ::: "memory");
#endif
}

// ---- totally-ordered float <-> u32 key (branch-free atomic max/min) -------
__device__ __forceinline__ unsigned fenc(float f) {
  const unsigned b = __float_as_uint(f);
  return b ^ ((unsigned)((int)b >> 31) | 0x80000000u);  // ashr+or+xor
}
__device__ __forceinline__ float fdec(unsigned e) {
  const unsigned x = (~(unsigned)((int)e >> 31)) | 0x80000000u;
  return __uint_as_float(e ^ x);
}
__device__ __forceinline__ unsigned umax2(unsigned a, unsigned b) { return a > b ? a : b; }
__device__ __forceinline__ unsigned umin2(unsigned a, unsigned b) { return a < b ? a : b; }

// ---------------- utility kernels ------------------------------------------
__global__ void fill_f(float* p, float v, int n) {
  int i = blockIdx.x * blockDim.x + threadIdx.x;
  if (i < n) p[i] = v;
}
__global__ void fill_u(unsigned* p, unsigned v, int n) {
  int i = blockIdx.x * blockDim.x + threadIdx.x;
  if (i < n) p[i] = v;
}

// convert W (fdim x dout, f32) -> WT (dout x kpad, f16), zero-padded cols
__global__ void convert_w_T(const float* __restrict__ W, _Float16* __restrict__ WT,
                            int fdim, int dout, int kpad) {
  int i = blockIdx.x * blockDim.x + threadIdx.x;
  if (i >= fdim * dout) return;
  int j = i / dout, c = i - j * dout;
  WT[(size_t)c * kpad + j] = (_Float16)W[i];
}

// ---------------- kNN: per-thread insertion-sorted top-20 ------------------
__global__ __launch_bounds__(256) void knn_kernel(const float* __restrict__ pts,
                                                  int* __restrict__ nbr) {
  __shared__ float xs[N_ * 3];
  __shared__ float x2s[N_];
  const int b   = blockIdx.x >> 2;
  const int tid = threadIdx.x;
  const float* base = pts + (size_t)b * N_ * 3;
  for (int i = tid; i < N_; i += 256) {
    float x = base[i * 3 + 0], y = base[i * 3 + 1], z = base[i * 3 + 2];
    xs[i * 3 + 0] = x; xs[i * 3 + 1] = y; xs[i * 3 + 2] = z;
    x2s[i] = x * x + y * y + z * z;
  }
  __syncthreads();
  const int p = (blockIdx.x & 3) * 256 + tid;
  const float px = xs[p * 3], py = xs[p * 3 + 1], pz = xs[p * 3 + 2], p2 = x2s[p];
  float bd[K_]; int bi[K_];
  for (int k = 0; k < K_; ++k) { bd[k] = INFINITY; bi[k] = 0; }
  for (int j = 0; j < N_; ++j) {
    if (j == p) continue;
    float d = p2 + x2s[j] - 2.f * (px * xs[j * 3] + py * xs[j * 3 + 1] + pz * xs[j * 3 + 2]);
    if (d < bd[K_ - 1]) {
      int k = K_ - 1;
      while (k > 0 && bd[k - 1] > d) { bd[k] = bd[k - 1]; bi[k] = bi[k - 1]; --k; }
      bd[k] = d; bi[k] = j;
    }
  }
  const int gp = b * N_ + p;
  for (int k = 0; k < K_; ++k) nbr[gp * K_ + k] = b * N_ + bi[k];
}

// ---------------- fused WMMA GEMM + BN-stats + group max/min ---------------
// A and B staged in LDS in WMMA fragment order (lane's 16 f16 contiguous ->
// ds_load_b128 x2 per fragment). Weights (WhT: dout x KPAD f16, transposed +
// zero padded) are copied via global_load_async_to_lds_b128.
template <int FDIM, int DOUT, int KPAD, int GROUP, bool EDGE, int DIN>
__global__ __launch_bounds__(256) void gemm_wmma(
    const float* __restrict__ X, int xld,
    const int* __restrict__ nbr,
    const _Float16* __restrict__ WhT,
    const float* __restrict__ bias,
    float* __restrict__ sums, float* __restrict__ sumsqs,
    unsigned* __restrict__ maxbuf, unsigned* __restrict__ minbuf) {
  constexpr int KCH     = KPAD / 32;                  // k-chunks in A
  constexpr int KSLAB   = (KPAD > 160) ? 160 : KPAD;  // B k-slab
  constexpr int NSLABS  = KPAD / KSLAB;
  constexpr int KCS     = KSLAB / 32;                 // k-chunks per B slab
  constexpr int NB      = 64;                         // n-block width
  constexpr int NBLOCKS = DOUT / NB;
  // float4-vectorizable staging: no padding and 8-aligned DIN (layers 2-4, glb)
  constexpr bool AL4 = (FDIM == KPAD) && (!EDGE || (DIN % 8 == 0));
  // lane's 8 consecutive rows can straddle a group boundary only if GROUP % 8
  constexpr bool SPLIT = (GROUP % 8) != 0;

  __shared__ __align__(16) _Float16 Afrag[64 * KPAD];      // 4*KCH tiles * 512
  __shared__ __align__(16) _Float16 Bfrag[KCS * 4 * 512];  // per-slab
  __shared__ float ssum[NB], ssq[NB];

  const int tid  = threadIdx.x;
  const int lane = tid & 31;
  const int wave = tid >> 5;
  const int mt   = wave & 3;    // m-tile 0..3
  const int nh   = wave >> 2;   // n-half 0..1
  const int rowBase = blockIdx.x * 64;

  // ---- stage A in fragment order: chunk = 8 consecutive f16 (one b128) ----
  for (int ci = tid; ci < 8 * KPAD; ci += 256) {
    const int tile = ci >> 6;           // 64 chunks per tile
    const int wi   = ci & 63;
    const int ln   = wi >> 1, hf = wi & 1;
    const int mts  = tile / KCH, kcs = tile - mts * KCH;
    const int ms   = ln & 15,  hs  = ln >> 4;
    const int row  = rowBase + mts * 16 + ms;
    const int j0   = kcs * 32 + (hf ? 16 : 0) + 8 * hs;   // e<8: 8h+e ; e>=8: 16+8h+e'
    const float* xp;
    const float* xq = nullptr;
    if constexpr (EDGE) {
      const int p  = row / K_;
      const int nq = nbr[row];
      xp = X + (size_t)p * xld;
      xq = X + (size_t)nq * xld;
    } else {
      xp = X + (size_t)row * xld;
    }
    union { _Float16 hv[8]; gv4i q4; } pk;
    if constexpr (AL4) {
      float f[8];
      if constexpr (EDGE) {
        if (j0 < DIN) {
          const float4 a0 = *(const float4*)(xp + j0);
          const float4 a1 = *(const float4*)(xp + j0 + 4);
          f[0] = a0.x; f[1] = a0.y; f[2] = a0.z; f[3] = a0.w;
          f[4] = a1.x; f[5] = a1.y; f[6] = a1.z; f[7] = a1.w;
        } else {
          const int jj = j0 - DIN;
          const float4 q0 = *(const float4*)(xq + jj);
          const float4 q1 = *(const float4*)(xq + jj + 4);
          const float4 p0 = *(const float4*)(xp + jj);
          const float4 p1 = *(const float4*)(xp + jj + 4);
          f[0] = q0.x - p0.x; f[1] = q0.y - p0.y; f[2] = q0.z - p0.z; f[3] = q0.w - p0.w;
          f[4] = q1.x - p1.x; f[5] = q1.y - p1.y; f[6] = q1.z - p1.z; f[7] = q1.w - p1.w;
        }
      } else {
        const float4 a0 = *(const float4*)(xp + j0);
        const float4 a1 = *(const float4*)(xp + j0 + 4);
        f[0] = a0.x; f[1] = a0.y; f[2] = a0.z; f[3] = a0.w;
        f[4] = a1.x; f[5] = a1.y; f[6] = a1.z; f[7] = a1.w;
      }
#pragma unroll
      for (int ee = 0; ee < 8; ++ee) pk.hv[ee] = (_Float16)f[ee];
    } else {
#pragma unroll
      for (int ee = 0; ee < 8; ++ee) {
        const int j = j0 + ee;
        float v = 0.f;
        if (j < FDIM) {
          if constexpr (EDGE) v = (j < DIN) ? xp[j] : (xq[j - DIN] - xp[j - DIN]);
          else                v = xp[j];
        }
        pk.hv[ee] = (_Float16)v;
      }
    }
    *(gv4i*)(Afrag + (tile << 9) + (ln << 4) + (hf << 3)) = pk.q4;
  }

  const int m = lane & 15;
  const int h = lane >> 4;

  for (int nblk = 0; nblk < NBLOCKS; ++nblk) {
    const int nb0 = nblk * NB;
    v8f acc0 = {}; v8f acc1 = {};
    if (tid < NB) { ssum[tid] = 0.f; ssq[tid] = 0.f; }

    for (int ks = 0; ks < NSLABS; ++ks) {
      __syncthreads();
      // ---- stage B slab in fragment order; 16B chunks contiguous in WhT ----
      for (int ci = tid; ci < KCS * 4 * 64; ci += 256) {
        const int tile = ci >> 6;       // kc*4 + nt
        const int wi   = ci & 63;
        const int ln   = wi >> 1, hf = wi & 1;
        const int kc   = tile >> 2, nt = tile & 3;
        const int mm   = ln & 15,  hh = ln >> 4;
        const int c    = nb0 + nt * 16 + mm;
        const int j    = ks * KSLAB + kc * 32 + 8 * hh + 16 * hf;
        const _Float16* src = WhT + (size_t)c * KPAD + j;
        _Float16*       dst = Bfrag + (tile << 9) + (ln << 4) + (hf << 3);
#if HAVE_ASYNC_B128
        __builtin_amdgcn_global_load_async_to_lds_b128(
            (gv4i*)src, lds_gv4i_ptr(dst), 0, 0);
#else
        *(gv4i*)dst = *(const gv4i*)src;
#endif
      }
#if HAVE_ASYNC_B128
      wait_async0();
#endif
      __syncthreads();

      for (int kc = 0; kc < KCS; ++kc) {
        union { v16h v; gv4i q[2]; } a, b0, b1;
        {
          const gv4i* p = (const gv4i*)(Afrag + ((mt * KCH + ks * KCS + kc) << 9) + (lane << 4));
          a.q[0] = p[0]; a.q[1] = p[1];
        }
        {
          const gv4i* p = (const gv4i*)(Bfrag + (((kc << 2) + nh * 2 + 0) << 9) + (lane << 4));
          b0.q[0] = p[0]; b0.q[1] = p[1];
        }
        {
          const gv4i* p = (const gv4i*)(Bfrag + (((kc << 2) + nh * 2 + 1) << 9) + (lane << 4));
          b1.q[0] = p[0]; b1.q[1] = p[1];
        }
        acc0 = __builtin_amdgcn_wmma_f32_16x16x32_f16(false, a.v, false, b0.v,
                                                      (short)0, acc0, false, false);
        acc1 = __builtin_amdgcn_wmma_f32_16x16x32_f16(false, a.v, false, b1.v,
                                                      (short)0, acc1, false, false);
      }
    }

    // ---- write out: bias, BN stats, branchless two-run group max/min ------
    // 8 consecutive rows per lane span at most 2 groups (GROUP >= 20).
#pragma unroll
    for (int t = 0; t < 2; ++t) {
      const v8f acc = t ? acc1 : acc0;
      const int colLocal = (nh * 2 + t) * 16 + m;
      const int col = nb0 + colLocal;
      const float bv = bias[col];
      float ps = 0.f, pq = 0.f;
      const int row0 = rowBase + mt * 16 + 8 * h;
      const int grp  = row0 / GROUP;
      unsigned max0 = 0u, min0 = 0xFFFFFFFFu;
      if constexpr (!SPLIT) {
        // rows never straddle a group boundary (GROUP % 8 == 0): single run
#pragma unroll
        for (int r = 0; r < 8; ++r) {
          const float y = acc[r] + bv;
          ps += y; pq += y * y;
          const unsigned key = fenc(y);
          max0 = umax2(max0, key);
          min0 = umin2(min0, key);
        }
        atomicMax(&maxbuf[(size_t)grp * DOUT + col], max0);
        atomicMin(&minbuf[(size_t)grp * DOUT + col], min0);
      } else {
        const int bnd = (grp + 1) * GROUP - row0;   // rows before group boundary
        unsigned max1 = 0u, min1 = 0xFFFFFFFFu;
#pragma unroll
        for (int r = 0; r < 8; ++r) {
          const float y = acc[r] + bv;
          ps += y; pq += y * y;
          const unsigned key = fenc(y);
          const bool in0 = (r < bnd);               // value-select, no branches
          max0 = umax2(max0, in0 ? key : 0u);
          min0 = umin2(min0, in0 ? key : 0xFFFFFFFFu);
          max1 = umax2(max1, in0 ? 0u : key);
          min1 = umin2(min1, in0 ? 0xFFFFFFFFu : key);
        }
        atomicMax(&maxbuf[(size_t)grp * DOUT + col], max0);
        atomicMin(&minbuf[(size_t)grp * DOUT + col], min0);
        if (bnd < 8) {
          atomicMax(&maxbuf[(size_t)(grp + 1) * DOUT + col], max1);
          atomicMin(&minbuf[(size_t)(grp + 1) * DOUT + col], min1);
        }
      }
      atomicAdd(&ssum[colLocal], ps);
      atomicAdd(&ssq[colLocal], pq);
    }
    __syncthreads();
    if (tid < NB) {
      atomicAdd(&sums[nb0 + tid],   ssum[tid]);
      atomicAdd(&sumsqs[nb0 + tid], ssq[tid]);
    }
  }
}

// ---------------- BN finalize + ReLU + (max-pool via monotone trick) -------
__global__ void bn_pool_finalize(const float* __restrict__ sums,
                                 const float* __restrict__ sumsqs, float inv_cnt,
                                 const float* __restrict__ g,
                                 const float* __restrict__ beta,
                                 const void* __restrict__ maxbuf,
                                 const void* __restrict__ minbuf, int encoded,
                                 float* __restrict__ out,
                                 int ngroups, int dout, int outld) {
  int i = blockIdx.x * blockDim.x + threadIdx.x;
  if (i >= ngroups * dout) return;
  const int grp = i / dout, c = i - grp * dout;
  const float mean = sums[c] * inv_cnt;
  const float var  = sumsqs[c] * inv_cnt - mean * mean;
  const float s = g[c] * rsqrtf(var + 1e-5f);
  const float t = beta[c] - mean * s;
  float v;
  if (encoded) {
    v = (s >= 0.f) ? fdec(((const unsigned*)maxbuf)[i])
                   : fdec(((const unsigned*)minbuf)[i]);
  } else {
    v = (s >= 0.f) ? ((const float*)maxbuf)[i] : ((const float*)minbuf)[i];
  }
  out[(size_t)grp * outld + c] = fmaxf(s * v + t, 0.f);
}

// ---------------- tiny fc GEMM (32 rows) + BN stats ------------------------
__global__ void fc_kernel(const float* __restrict__ X, const float* __restrict__ W,
                          const float* __restrict__ b, float* __restrict__ Y,
                          float* __restrict__ sums, float* __restrict__ sumsqs,
                          int rows, int din, int dout) {
  int i = blockIdx.x * blockDim.x + threadIdx.x;
  if (i >= rows * dout) return;
  const int r = i / dout, c = i - r * dout;
  float acc = b[c];
  for (int j = 0; j < din; ++j) acc += X[(size_t)r * din + j] * W[(size_t)j * dout + c];
  Y[i] = acc;
  atomicAdd(&sums[c], acc);
  atomicAdd(&sumsqs[c], acc * acc);
}

// ---------------- classifier head: logits + log_softmax --------------------
__global__ void head_out(const float* __restrict__ F, const float* __restrict__ W,
                         const float* __restrict__ b, float* __restrict__ out) {
  __shared__ float logits[40];
  __shared__ float red[2];
  const int r = blockIdx.x, t = threadIdx.x;
  if (t < 40) {
    float acc = b[t];
    for (int j = 0; j < 256; ++j) acc += F[r * 256 + j] * W[j * 40 + t];
    logits[t] = acc;
  }
  __syncthreads();
  if (t == 0) {
    float mx = -INFINITY;
    for (int c = 0; c < 40; ++c) mx = fmaxf(mx, logits[c]);
    float s = 0.f;
    for (int c = 0; c < 40; ++c) s += __expf(logits[c] - mx);
    red[0] = mx; red[1] = __logf(s);
  }
  __syncthreads();
  if (t < 40) out[r * 40 + t] = logits[t] - red[0] - red[1];
}

// ===========================================================================
extern "C" void kernel_launch(void* const* d_in, const int* in_sizes, int n_in,
                              void* d_out, int out_size, void* d_ws, size_t ws_size,
                              hipStream_t stream) {
  const float* pts = (const float*)d_in[0];
  const float* eW[4]  = {(const float*)d_in[2],  (const float*)d_in[6],
                         (const float*)d_in[10], (const float*)d_in[14]};
  const float* eb[4]  = {(const float*)d_in[3],  (const float*)d_in[7],
                         (const float*)d_in[11], (const float*)d_in[15]};
  const float* eg[4]  = {(const float*)d_in[4],  (const float*)d_in[8],
                         (const float*)d_in[12], (const float*)d_in[16]};
  const float* ebt[4] = {(const float*)d_in[5],  (const float*)d_in[9],
                         (const float*)d_in[13], (const float*)d_in[17]};
  const float* gWf = (const float*)d_in[18];
  const float* gb  = (const float*)d_in[19];
  const float* gg  = (const float*)d_in[20];
  const float* gbt = (const float*)d_in[21];
  const float* f1W = (const float*)d_in[22];
  const float* f1b = (const float*)d_in[23];
  const float* f1g = (const float*)d_in[24];
  const float* f1t = (const float*)d_in[25];
  const float* f2W = (const float*)d_in[26];
  const float* f2b = (const float*)d_in[27];
  const float* f2g = (const float*)d_in[28];
  const float* f2t = (const float*)d_in[29];
  const float* oW  = (const float*)d_in[30];
  const float* ob  = (const float*)d_in[31];
  float* out = (float*)d_out;

  // ---- workspace carve-out ----
  char* ws = (char*)d_ws;
  size_t off = 0;
  auto take = [&](size_t bytes) -> void* {
    void* p = ws + off;
    off = (off + bytes + 255) & ~(size_t)255;
    return p;
  };
  int*      nbr  = (int*)take((size_t)E_ * 4);
  float*    cat  = (float*)take((size_t)P_ * CATD * 4);
  _Float16* wh0T = (_Float16*)take((size_t)64 * 32 * 2);      // dout x KPAD
  _Float16* wh1T = (_Float16*)take((size_t)64 * 128 * 2);
  _Float16* wh2T = (_Float16*)take((size_t)64 * 128 * 2);
  _Float16* wh3T = (_Float16*)take((size_t)128 * 128 * 2);
  _Float16* whgT = (_Float16*)take((size_t)1024 * 320 * 2);
  float*    sums = (float*)take(1024 * 4);
  float*    sqs  = (float*)take(1024 * 4);
  unsigned* maxb = (unsigned*)take((size_t)P_ * 128 * 4);
  unsigned* minb = (unsigned*)take((size_t)P_ * 128 * 4);
  float*    pooled = (float*)take((size_t)32 * 1024 * 4);
  float*    f1r  = (float*)take((size_t)32 * 512 * 4);
  float*    f1   = (float*)take((size_t)32 * 512 * 4);
  float*    f2r  = (float*)take((size_t)32 * 256 * 4);
  float*    f2   = (float*)take((size_t)32 * 256 * 4);
  (void)in_sizes; (void)n_in; (void)out_size; (void)ws_size;

  auto cdiv = [](int a, int b) { return (a + b - 1) / b; };

  // ---- kNN ----
  knn_kernel<<<B_ * (N_ / 256), 256, 0, stream>>>(pts, nbr);

  // ---- weight prep: zero-pad + transpose-convert to f16 ----
  fill_f<<<cdiv(64 * 32 / 2, 256), 256, 0, stream>>>((float*)wh0T, 0.f, 64 * 32 / 2);
  fill_f<<<cdiv(64 * 128 / 2, 256), 256, 0, stream>>>((float*)wh1T, 0.f, 64 * 128 / 2);
  fill_f<<<cdiv(64 * 128 / 2, 256), 256, 0, stream>>>((float*)wh2T, 0.f, 64 * 128 / 2);
  fill_f<<<cdiv(128 * 128 / 2, 256), 256, 0, stream>>>((float*)wh3T, 0.f, 128 * 128 / 2);
  convert_w_T<<<cdiv(6 * 64, 256), 256, 0, stream>>>(eW[0], wh0T, 6, 64, 32);
  convert_w_T<<<cdiv(128 * 64, 256), 256, 0, stream>>>(eW[1], wh1T, 128, 64, 128);
  convert_w_T<<<cdiv(128 * 64, 256), 256, 0, stream>>>(eW[2], wh2T, 128, 64, 128);
  convert_w_T<<<cdiv(128 * 128, 256), 256, 0, stream>>>(eW[3], wh3T, 128, 128, 128);
  convert_w_T<<<cdiv(320 * 1024, 256), 256, 0, stream>>>(gWf, whgT, 320, 1024, 320);

  const float EINV = 1.f / (float)E_;
  const float PINV = 1.f / (float)P_;

  // ---- EdgeConv layer 1: pts(3) -> cat[:, 0:64] ----
  fill_f<<<4, 256, 0, stream>>>(sums, 0.f, 1024);
  fill_f<<<4, 256, 0, stream>>>(sqs, 0.f, 1024);
  fill_u<<<cdiv(P_ * 64, 256), 256, 0, stream>>>(maxb, 0u, P_ * 64);
  fill_u<<<cdiv(P_ * 64, 256), 256, 0, stream>>>(minb, 0xFFFFFFFFu, P_ * 64);
  gemm_wmma<6, 64, 32, K_, true, 3><<<E_ / 64, 256, 0, stream>>>(
      pts, 3, nbr, wh0T, eb[0], sums, sqs, maxb, minb);
  bn_pool_finalize<<<cdiv(P_ * 64, 256), 256, 0, stream>>>(
      sums, sqs, EINV, eg[0], ebt[0], maxb, minb, 1, cat + 0, P_, 64, CATD);

  // ---- EdgeConv layer 2 ----
  fill_f<<<4, 256, 0, stream>>>(sums, 0.f, 1024);
  fill_f<<<4, 256, 0, stream>>>(sqs, 0.f, 1024);
  fill_u<<<cdiv(P_ * 64, 256), 256, 0, stream>>>(maxb, 0u, P_ * 64);
  fill_u<<<cdiv(P_ * 64, 256), 256, 0, stream>>>(minb, 0xFFFFFFFFu, P_ * 64);
  gemm_wmma<128, 64, 128, K_, true, 64><<<E_ / 64, 256, 0, stream>>>(
      cat + 0, CATD, nbr, wh1T, eb[1], sums, sqs, maxb, minb);
  bn_pool_finalize<<<cdiv(P_ * 64, 256), 256, 0, stream>>>(
      sums, sqs, EINV, eg[1], ebt[1], maxb, minb, 1, cat + 64, P_, 64, CATD);

  // ---- EdgeConv layer 3 ----
  fill_f<<<4, 256, 0, stream>>>(sums, 0.f, 1024);
  fill_f<<<4, 256, 0, stream>>>(sqs, 0.f, 1024);
  fill_u<<<cdiv(P_ * 64, 256), 256, 0, stream>>>(maxb, 0u, P_ * 64);
  fill_u<<<cdiv(P_ * 64, 256), 256, 0, stream>>>(minb, 0xFFFFFFFFu, P_ * 64);
  gemm_wmma<128, 64, 128, K_, true, 64><<<E_ / 64, 256, 0, stream>>>(
      cat + 64, CATD, nbr, wh2T, eb[2], sums, sqs, maxb, minb);
  bn_pool_finalize<<<cdiv(P_ * 64, 256), 256, 0, stream>>>(
      sums, sqs, EINV, eg[2], ebt[2], maxb, minb, 1, cat + 128, P_, 64, CATD);

  // ---- EdgeConv layer 4 ----
  fill_f<<<4, 256, 0, stream>>>(sums, 0.f, 1024);
  fill_f<<<4, 256, 0, stream>>>(sqs, 0.f, 1024);
  fill_u<<<cdiv(P_ * 128, 256), 256, 0, stream>>>(maxb, 0u, P_ * 128);
  fill_u<<<cdiv(P_ * 128, 256), 256, 0, stream>>>(minb, 0xFFFFFFFFu, P_ * 128);
  gemm_wmma<128, 128, 128, K_, true, 64><<<E_ / 64, 256, 0, stream>>>(
      cat + 128, CATD, nbr, wh3T, eb[3], sums, sqs, maxb, minb);
  bn_pool_finalize<<<cdiv(P_ * 128, 256), 256, 0, stream>>>(
      sums, sqs, EINV, eg[3], ebt[3], maxb, minb, 1, cat + 192, P_, 128, CATD);

  // ---- global layer: cat(320) -> 1024, BN, ReLU, per-batch max-pool ----
  fill_f<<<4, 256, 0, stream>>>(sums, 0.f, 1024);
  fill_f<<<4, 256, 0, stream>>>(sqs, 0.f, 1024);
  fill_u<<<cdiv(B_ * 1024, 256), 256, 0, stream>>>(maxb, 0u, B_ * 1024);
  fill_u<<<cdiv(B_ * 1024, 256), 256, 0, stream>>>(minb, 0xFFFFFFFFu, B_ * 1024);
  gemm_wmma<320, 1024, 320, N_, false, 0><<<P_ / 64, 256, 0, stream>>>(
      cat, CATD, nullptr, whgT, gb, sums, sqs, maxb, minb);
  bn_pool_finalize<<<cdiv(B_ * 1024, 256), 256, 0, stream>>>(
      sums, sqs, PINV, gg, gbt, maxb, minb, 1, pooled, B_, 1024, 1024);

  // ---- fc1 ----
  fill_f<<<4, 256, 0, stream>>>(sums, 0.f, 1024);
  fill_f<<<4, 256, 0, stream>>>(sqs, 0.f, 1024);
  fc_kernel<<<cdiv(32 * 512, 256), 256, 0, stream>>>(pooled, f1W, f1b, f1r,
                                                     sums, sqs, 32, 1024, 512);
  bn_pool_finalize<<<cdiv(32 * 512, 256), 256, 0, stream>>>(
      sums, sqs, 1.f / 32.f, f1g, f1t, f1r, f1r, 0, f1, 32, 512, 512);

  // ---- fc2 ----
  fill_f<<<4, 256, 0, stream>>>(sums, 0.f, 1024);
  fill_f<<<4, 256, 0, stream>>>(sqs, 0.f, 1024);
  fc_kernel<<<cdiv(32 * 256, 256), 256, 0, stream>>>(f1, f2W, f2b, f2r,
                                                     sums, sqs, 32, 512, 256);
  bn_pool_finalize<<<cdiv(32 * 256, 256), 256, 0, stream>>>(
      sums, sqs, 1.f / 32.f, f2g, f2t, f2r, f2r, 0, f2, 32, 256, 256);

  // ---- output head ----
  head_out<<<32, 64, 0, stream>>>(f2, oW, ob, out);
}
